// EEGConvNetMiniV2Attention_27049704030295
// MI455X (gfx1250) — compile-verified, matrix operand order checked by reading, not run on hardware
//
#include <hip/hip_runtime.h>
#include <cstdint>

#define N_NODES 50000
#define N_EDGES 800000
#define N_TOT   (N_EDGES + N_NODES)   // edges + self loops
#define D_IN    128
#define NUM_GRAPHS 1000
#define H1 32
#define H2 64
#define H3 128
#define POOL_PAD 1008                 // 63 * 16
#define CDIV(a,b) (((a)+(b)-1)/(b))

typedef __attribute__((ext_vector_type(16))) _Float16 v16h;
typedef __attribute__((ext_vector_type(8)))  _Float16 v8h;
typedef __attribute__((ext_vector_type(8)))  float    v8f;

static __device__ __forceinline__ float lrelu(float v, float s) { return v > 0.f ? v : s * v; }
// order-preserving float<->int map for atomicMax on floats
static __device__ __forceinline__ int   fmap(float f) { int i = __float_as_int(f); return i >= 0 ? i : (i ^ 0x7fffffff); }
static __device__ __forceinline__ float funmap(int i) { return __int_as_float(i >= 0 ? i : (i ^ 0x7fffffff)); }

// ---------------- utility kernels ----------------
__global__ void fill_f32(float* p, float v, long long n) {
    long long i = (long long)blockIdx.x * blockDim.x + threadIdx.x;
    if (i < n) p[i] = v;
}
__global__ void fill_i32(int* p, int v, int n) {
    int i = blockIdx.x * blockDim.x + threadIdx.x;
    if (i < n) p[i] = v;
}
__global__ void cvt_f16(const float* in, _Float16* out, long long n) {
    long long i = (long long)blockIdx.x * blockDim.x + threadIdx.x;
    if (i < n) out[i] = (_Float16)in[i];
}
// W is [K x M] row-major; write Wt as [M x K] so B-fragment loads are contiguous
__global__ void cvt_f16_t(const float* W, _Float16* Wt, int K, int M) {
    int i = blockIdx.x * blockDim.x + threadIdx.x;
    if (i >= K * M) return;
    int k = i / M, m = i % M;
    Wt[(size_t)m * K + k] = (_Float16)W[i];
}

// ---------------- WMMA GEMM: C[N x M] = A[N x K] * Bt[M x K]^T ----------------
// One wave per 16x16 output tile. K multiple of 32, M multiple of 16, N multiple of 16.
__global__ void __launch_bounds__(32) wmma_gemm_f16(const _Float16* __restrict__ A,
                                                    const _Float16* __restrict__ Bt,
                                                    float* __restrict__ C,
                                                    int K, int M) {
    const int tn   = blockIdx.x;        // row tile
    const int tm   = blockIdx.y;        // col tile
    const int lane = threadIdx.x;       // 0..31
    const int half = lane >> 4;         // 0: K-low half, 1: K-high half
    const int l    = lane & 15;

    const _Float16* Arow = A  + (size_t)(tn * 16 + l) * K;   // A row for this lane
    const _Float16* Bcol = Bt + (size_t)(tm * 16 + l) * K;   // B column for this lane

    v8f acc = {};
    for (int k0 = 0; k0 < K; k0 += 32) {
        // 16-bit 16x32 A layout: lanes 0-15 hold K = k0+[0..7] and k0+[16..23],
        // lanes 16-31 hold K = k0+[8..15] and k0+[24..31]  (B mirrored over columns)
        const int kb = k0 + half * 8;
        v8h alo = *(const v8h*)(Arow + kb);
        v8h ahi = *(const v8h*)(Arow + kb + 16);
        v8h blo = *(const v8h*)(Bcol + kb);
        v8h bhi = *(const v8h*)(Bcol + kb + 16);
        v16h a, b;
#pragma unroll
        for (int i = 0; i < 8; ++i) {
            a[i] = alo[i]; a[i + 8] = ahi[i];
            b[i] = blo[i]; b[i + 8] = bhi[i];
        }
        acc = __builtin_amdgcn_wmma_f32_16x16x32_f16(
            /*neg_a=*/false, a, /*neg_b=*/false, b,
            /*c_mod=*/(short)0, acc, /*reuse_a=*/false, /*reuse_b=*/false);
    }
    // C/D layout: VGPR r -> tile row (r + half*8), column = lane&15
    const int r0 = half * 8;
#pragma unroll
    for (int r = 0; r < 8; ++r)
        C[(size_t)(tn * 16 + r0 + r) * M + tm * 16 + l] = acc[r];
}

// ---------------- GAT attention pipeline ----------------
__global__ void node_logits(const float* __restrict__ h, const float* __restrict__ as,
                            const float* __restrict__ ad, float* esrc, float* edst, int M) {
    int i = blockIdx.x * blockDim.x + threadIdx.x;
    if (i >= N_NODES) return;
    float s = 0.f, d = 0.f;
    const float* hp = h + (size_t)i * M;
    for (int c = 0; c < M; ++c) { float v = hp[c]; s += v * as[c]; d += v * ad[c]; }
    esrc[i] = s; edst[i] = d;
}

__global__ void edge_logits_max(const int* __restrict__ ei, const float* __restrict__ esrc,
                                const float* __restrict__ edst, float* eedge, int* emax) {
    int t = blockIdx.x * blockDim.x + threadIdx.x;
    if (t >= N_TOT) return;
    int s, d;
    if (t < N_EDGES) { s = ei[t]; d = ei[N_EDGES + t]; } else { s = d = t - N_EDGES; }
    float e = lrelu(esrc[s] + edst[d], 0.2f);
    eedge[t] = e;
    atomicMax(&emax[d], fmap(e));
}

__global__ void edge_exp_sum(const int* __restrict__ ei, float* eedge,
                             const int* __restrict__ emax, float* denom) {
    int t = blockIdx.x * blockDim.x + threadIdx.x;
    if (t >= N_TOT) return;
    int d = (t < N_EDGES) ? ei[N_EDGES + t] : (t - N_EDGES);
    float ee = __expf(eedge[t] - funmap(emax[d]));
    eedge[t] = ee;                       // overwrite logit with exp value
    atomicAdd(&denom[d], ee);
}

__global__ void edge_aggregate(const int* __restrict__ ei, const float* __restrict__ eedge,
                               const float* __restrict__ denom, const float* __restrict__ h,
                               float* out, int M) {
    const int chunks = M >> 2;
    long long t = (long long)blockIdx.x * blockDim.x + threadIdx.x;
    if (t >= (long long)N_TOT * chunks) return;
    int e = (int)(t / chunks);
    int q = ((int)(t % chunks)) << 2;
    int s, d;
    if (e < N_EDGES) { s = ei[e]; d = ei[N_EDGES + e]; } else { s = d = e - N_EDGES; }
    float alpha = eedge[e] / (denom[d] + 1e-16f);
    const float4 hv = *(const float4*)(h + (size_t)s * M + q);
    float* op = out + (size_t)d * M + q;
    atomicAdd(op + 0, alpha * hv.x);
    atomicAdd(op + 1, alpha * hv.y);
    atomicAdd(op + 2, alpha * hv.z);
    atomicAdd(op + 3, alpha * hv.w);
}

// ---------------- BatchNorm (training stats) + leaky relu ----------------
__global__ void bn_stats(const float* __restrict__ x, const float* __restrict__ bias,
                         float* mean, float* rstd, int M) {
    __shared__ float s_sum[256], s_sq[256];
    const int c = blockIdx.x;
    float sum = 0.f, sq = 0.f;
    for (int i = threadIdx.x; i < N_NODES; i += blockDim.x) {
        float v = x[(size_t)i * M + c] + bias[c];
        sum += v; sq += v * v;
    }
    s_sum[threadIdx.x] = sum; s_sq[threadIdx.x] = sq;
    __syncthreads();
    for (int s = 128; s > 0; s >>= 1) {
        if (threadIdx.x < s) {
            s_sum[threadIdx.x] += s_sum[threadIdx.x + s];
            s_sq[threadIdx.x]  += s_sq[threadIdx.x + s];
        }
        __syncthreads();
    }
    if (threadIdx.x == 0) {
        float mu = s_sum[0] / (float)N_NODES;
        float var = s_sq[0] / (float)N_NODES - mu * mu;
        mean[c] = mu;
        rstd[c] = rsqrtf(var + 1e-5f);
    }
}

__global__ void bn_lrelu(float* x, const float* __restrict__ bias, const float* __restrict__ mean,
                         const float* __restrict__ rstd, const float* __restrict__ gamma,
                         const float* __restrict__ beta, int M) {
    long long i = (long long)blockIdx.x * blockDim.x + threadIdx.x;
    if (i >= (long long)N_NODES * M) return;
    int c = (int)(i % M);
    float v = x[i] + bias[c];
    v = (v - mean[c]) * rstd[c] * gamma[c] + beta[c];
    x[i] = lrelu(v, 0.01f);
}

// ---------------- pooling + MLP head ----------------
__global__ void pool_kernel(const float* __restrict__ h, const int* __restrict__ batch, float* pool) {
    int t = blockIdx.x * blockDim.x + threadIdx.x;       // N_NODES * (H3/4)
    if (t >= N_NODES * (H3 / 4)) return;
    int i = t >> 5;                                       // H3/4 == 32 chunks per node
    int q = (t & 31) << 2;
    int g = batch[i];
    const float4 hv = *(const float4*)(h + (size_t)i * H3 + q);
    float* pp = pool + (size_t)g * H3 + q;
    atomicAdd(pp + 0, hv.x);
    atomicAdd(pp + 1, hv.y);
    atomicAdd(pp + 2, hv.z);
    atomicAdd(pp + 3, hv.w);
}

__global__ void bias_lrelu(float* x, const float* __restrict__ b, long long n, int M) {
    long long i = (long long)blockIdx.x * blockDim.x + threadIdx.x;
    if (i >= n) return;
    x[i] = lrelu(x[i] + b[(int)(i % M)], 0.01f);
}

__global__ void fc3_kernel(const float* __restrict__ x, const float* __restrict__ w,
                           const float* __restrict__ b, float* out) {
    int t = blockIdx.x * blockDim.x + threadIdx.x;
    if (t >= NUM_GRAPHS * 2) return;
    int r = t >> 1, c = t & 1;
    float acc = b[c];
    const float* xp = x + (size_t)r * H2;
    for (int k = 0; k < H2; ++k) acc += xp[k] * w[k * 2 + c];
    out[t] = lrelu(acc, 0.01f);
}

// ---------------- host launch ----------------
extern "C" void kernel_launch(void* const* d_in, const int* in_sizes, int n_in,
                              void* d_out, int out_size, void* d_ws, size_t ws_size,
                              hipStream_t stream) {
    const float* x     = (const float*)d_in[0];
    const int*   ei    = (const int*)d_in[2];
    const int*   batch = (const int*)d_in[3];
    const float* W[3]  = {(const float*)d_in[4],  (const float*)d_in[8],  (const float*)d_in[12]};
    const float* As[3] = {(const float*)d_in[5],  (const float*)d_in[9],  (const float*)d_in[13]};
    const float* Ad[3] = {(const float*)d_in[6],  (const float*)d_in[10], (const float*)d_in[14]};
    const float* Bb[3] = {(const float*)d_in[7],  (const float*)d_in[11], (const float*)d_in[15]};
    const float* Gm[3] = {(const float*)d_in[16], (const float*)d_in[18], (const float*)d_in[20]};
    const float* Be[3] = {(const float*)d_in[17], (const float*)d_in[19], (const float*)d_in[21]};
    const float* fc1_w = (const float*)d_in[22];
    const float* fc1_b = (const float*)d_in[23];
    const float* fc2_w = (const float*)d_in[24];
    const float* fc2_b = (const float*)d_in[25];
    const float* fc3_w = (const float*)d_in[26];
    const float* fc3_b = (const float*)d_in[27];
    float* out = (float*)d_out;

    // workspace layout (256B aligned slices)
    uint8_t* wsb = (uint8_t*)d_ws;
    size_t off = 0;
    auto take = [&](size_t bytes) -> void* {
        void* p = wsb + off;
        off = (off + bytes + 255) & ~(size_t)255;
        return p;
    };
    _Float16* XH    = (_Float16*)take((size_t)N_NODES * 128 * 2);
    _Float16* WHT   = (_Float16*)take((size_t)128 * 128 * 2);
    float*    HBUF  = (float*)take((size_t)N_NODES * 128 * 4);
    float*    OUTB  = (float*)take((size_t)N_NODES * 128 * 4);
    float*    ESRC  = (float*)take((size_t)N_NODES * 4);
    float*    EDST  = (float*)take((size_t)N_NODES * 4);
    float*    EEDGE = (float*)take((size_t)N_TOT * 4);
    int*      EMAX  = (int*)take((size_t)N_NODES * 4);
    float*    DENOM = (float*)take((size_t)N_NODES * 4);
    float*    MEAN  = (float*)take(128 * 4);
    float*    RSTD  = (float*)take(128 * 4);
    float*    POOL  = (float*)take((size_t)POOL_PAD * 128 * 4);
    _Float16* POOLH = (_Float16*)take((size_t)POOL_PAD * 128 * 2);
    float*    FC1O  = (float*)take((size_t)POOL_PAD * 128 * 4);
    float*    FC2O  = (float*)take((size_t)POOL_PAD * 64 * 4);

    const int Ks[3] = {D_IN, H1, H2};
    const int Ms[3] = {H1, H2, H3};
    const float* in_f32 = x;

    for (int L = 0; L < 3; ++L) {
        const int K = Ks[L], M = Ms[L];
        const long long nIn = (long long)N_NODES * K;
        const long long nOut = (long long)N_NODES * M;

        cvt_f16<<<CDIV(nIn, 256), 256, 0, stream>>>(in_f32, XH, nIn);
        cvt_f16_t<<<CDIV(K * M, 256), 256, 0, stream>>>(W[L], WHT, K, M);
        wmma_gemm_f16<<<dim3(N_NODES / 16, M / 16), 32, 0, stream>>>(XH, WHT, HBUF, K, M);

        node_logits<<<CDIV(N_NODES, 256), 256, 0, stream>>>(HBUF, As[L], Ad[L], ESRC, EDST, M);

        fill_i32<<<CDIV(N_NODES, 256), 256, 0, stream>>>(EMAX, (int)0x80000000, N_NODES);
        fill_f32<<<CDIV(N_NODES, 256), 256, 0, stream>>>(DENOM, 0.f, N_NODES);
        fill_f32<<<CDIV(nOut, 256), 256, 0, stream>>>(OUTB, 0.f, nOut);

        edge_logits_max<<<CDIV(N_TOT, 256), 256, 0, stream>>>(ei, ESRC, EDST, EEDGE, EMAX);
        edge_exp_sum<<<CDIV(N_TOT, 256), 256, 0, stream>>>(ei, EEDGE, EMAX, DENOM);
        {
            long long nAgg = (long long)N_TOT * (M >> 2);
            edge_aggregate<<<CDIV(nAgg, 256), 256, 0, stream>>>(ei, EEDGE, DENOM, HBUF, OUTB, M);
        }

        bn_stats<<<M, 256, 0, stream>>>(OUTB, Bb[L], MEAN, RSTD, M);
        bn_lrelu<<<CDIV(nOut, 256), 256, 0, stream>>>(OUTB, Bb[L], MEAN, RSTD, Gm[L], Be[L], M);

        in_f32 = OUTB;
    }

    // global_add_pool into padded [1008 x 128] (tail rows stay zero for WMMA)
    fill_f32<<<CDIV((long long)POOL_PAD * 128, 256), 256, 0, stream>>>(POOL, 0.f, (long long)POOL_PAD * 128);
    pool_kernel<<<CDIV(N_NODES * (H3 / 4), 256), 256, 0, stream>>>(OUTB, batch, POOL);

    // fc1: [1008 x 128] @ [128 x 128]
    cvt_f16<<<CDIV((long long)POOL_PAD * 128, 256), 256, 0, stream>>>(POOL, POOLH, (long long)POOL_PAD * 128);
    cvt_f16_t<<<CDIV(128 * 128, 256), 256, 0, stream>>>(fc1_w, WHT, 128, 128);
    wmma_gemm_f16<<<dim3(POOL_PAD / 16, 128 / 16), 32, 0, stream>>>(POOLH, WHT, FC1O, 128, 128);
    bias_lrelu<<<CDIV((long long)POOL_PAD * 128, 256), 256, 0, stream>>>(FC1O, fc1_b, (long long)POOL_PAD * 128, 128);

    // fc2: [1008 x 128] @ [128 x 64]
    cvt_f16<<<CDIV((long long)POOL_PAD * 128, 256), 256, 0, stream>>>(FC1O, POOLH, (long long)POOL_PAD * 128);
    cvt_f16_t<<<CDIV(128 * 64, 256), 256, 0, stream>>>(fc2_w, WHT, 128, 64);
    wmma_gemm_f16<<<dim3(POOL_PAD / 16, 64 / 16), 32, 0, stream>>>(POOLH, WHT, FC2O, 128, 64);
    bias_lrelu<<<CDIV((long long)POOL_PAD * 64, 256), 256, 0, stream>>>(FC2O, fc2_b, (long long)POOL_PAD * 64, 64);

    // fc3: [1000 x 64] @ [64 x 2] -> d_out (scalar; M=2 not WMMA-tileable)
    fc3_kernel<<<CDIV(NUM_GRAPHS * 2, 256), 256, 0, stream>>>(FC2O, fc3_w, fc3_b, out);

    (void)in_sizes; (void)n_in; (void)out_size; (void)ws_size;
}
